// HyperbolicAgg_57638461112977
// MI455X (gfx1250) — matrix operand order repaired and checked by reading, not compiled, not run on hardware
//
#include <hip/hip_runtime.h>
#include <hip/hip_bf16.h>
#include <math.h>

#define D   128
#define DS  127
#define EPSF 1e-7f

typedef __attribute__((ext_vector_type(16))) _Float16 v16h;
typedef __attribute__((ext_vector_type(8)))  _Float16 v8h;
typedef __attribute__((ext_vector_type(4)))  _Float16 v4h;
typedef __attribute__((ext_vector_type(8)))  float    v8f;

__device__ __forceinline__ float wave_reduce_sum(float v) {
#pragma unroll
  for (int off = 16; off >= 1; off >>= 1) v += __shfl_xor(v, off, 32);
  return v;
}

// ---- prep: W (127x127 f32, row-major) -> WhT (128x128 f16), WhT[k][n] = W[n][k], zero-padded
__global__ void k_prep_weight(const float* __restrict__ w, _Float16* __restrict__ WhT) {
  int idx = blockIdx.x * blockDim.x + threadIdx.x;   // 0..16383
  int k = idx >> 7;
  int n = idx & 127;
  float val = (k < DS && n < DS) ? w[n * DS + k] : 0.0f;
  WhT[idx] = (_Float16)val;
}

// ---- logmap0: x (N x 128 f32) -> v_h (N x 128 f16), v_h[i][j] = scale_i * x[i][1+j], col 127 = 0
__global__ void k_logmap(const float* __restrict__ x, _Float16* __restrict__ vh, int N) {
  int row = blockIdx.x * 8 + (threadIdx.x >> 5);
  if (row >= N) return;
  int lane = threadIdx.x & 31;
  const float4 xi = *(const float4*)(x + (size_t)row * D + lane * 4);
  float ss = xi.y * xi.y + xi.z * xi.z + xi.w * xi.w;
  if (lane != 0) ss += xi.x * xi.x;          // spatial norm excludes element 0
  ss = wave_reduce_sum(ss);
  float x0 = __shfl(xi.x, 0, 32);
  float y  = fmaxf(x0, 1.0f + EPSF);
  float dd = logf(y + sqrtf(y * y - 1.0f));  // arccosh
  float sn = fmaxf(sqrtf(ss), EPSF);
  float scale = dd / sn;
  float nx = __shfl(xi.x, (lane + 1) & 31, 32);   // x[i][4*lane+4]
  v4h o;
  o[0] = (_Float16)(scale * xi.y);
  o[1] = (_Float16)(scale * xi.z);
  o[2] = (_Float16)(scale * xi.w);
  o[3] = (lane == 31) ? (_Float16)0.0f : (_Float16)(scale * nx);
  *(v4h*)(vh + (size_t)row * D + lane * 4) = o;
}

// ---- WMMA fragment loaders (16-bit, 16x16x32) ----
// A 16x32: lanes 0-15 row M=L, halves = K {0..7,16..23}; lanes 16-31 same rows, K {8..15,24..31}
__device__ __forceinline__ v16h load_fragA(const _Float16* __restrict__ base, int lane, int ks) {
  int r  = lane & 15;
  int kb = ks * 32 + ((lane >> 4) << 3);
  const _Float16* p = base + (size_t)r * D + kb;
  v8h lo = *(const v8h*)p;
  v8h hi = *(const v8h*)(p + 16);
  return __builtin_shufflevector(lo, hi, 0,1,2,3,4,5,6,7,8,9,10,11,12,13,14,15);
}
// B 32x16: lane L holds column N=n0+(L&15), K {0..7,16..23} (lanes 0-15) / {8..15,24..31} (lanes 16-31)
__device__ __forceinline__ v16h load_fragB(const _Float16* __restrict__ WhT, int lane, int ks, int n0) {
  const _Float16* p = WhT + (size_t)(ks * 32 + ((lane >> 4) << 3)) * D + n0 + (lane & 15);
  v16h b;
#pragma unroll
  for (int i = 0; i < 8; ++i) {
    b[i]     = p[(size_t)i * D];
    b[i + 8] = p[(size_t)(16 + i) * D];
  }
  return b;
}

// ---- GEMM: s = v_h @ WhT  (one wave per 16 rows; 4 K-steps x 8 N-tiles = 32 WMMAs)
__global__ void k_gemm(const _Float16* __restrict__ vh, const _Float16* __restrict__ WhT,
                       float* __restrict__ s, int ntiles) {
  int tile = blockIdx.x * 8 + (threadIdx.x >> 5);
  if (tile >= ntiles) return;
  int lane = threadIdx.x & 31;
  int m0 = tile * 16;
  v8f acc[8] = {};
  const _Float16* abase = vh + (size_t)m0 * D;
#pragma unroll
  for (int ks = 0; ks < 4; ++ks) {
    v16h a = load_fragA(abase, lane, ks);
#pragma unroll
    for (int n = 0; n < 8; ++n) {
      v16h b = load_fragB(WhT, lane, ks, n * 16);
      acc[n] = __builtin_amdgcn_wmma_f32_16x16x32_f16(
          false, a, false, b, (short)0, acc[n], false, false);
    }
  }
  // C 16x16 f32: VGPR r, lanes 0-15 -> (M=r, N=lane); lanes 16-31 -> (M=r+8, N=lane-16)
  int rr = (lane >> 4) << 3;
  int cc = lane & 15;
#pragma unroll
  for (int n = 0; n < 8; ++n) {
#pragma unroll
    for (int r = 0; r < 8; ++r) {
      s[(size_t)(m0 + r + rr) * D + n * 16 + cc] = acc[n][r];
    }
  }
}

// ---- expmap0 + proj: s (+bias) -> h (N x 128), h[0]=sqrt(1+||sp'||^2), h[1+j]=sinh(un)/un * sp[j]
__global__ void k_expmap(const float* __restrict__ s, const float* __restrict__ bias,
                         float* __restrict__ h, int N) {
  int row = blockIdx.x * 8 + (threadIdx.x >> 5);
  if (row >= N) return;
  int lane = threadIdx.x & 31;
  float4 sp = *(const float4*)(s + (size_t)row * D + lane * 4);
  int j0 = lane * 4;
  sp.x += (j0 + 0 < DS) ? bias[j0 + 0] : 0.0f;
  sp.y += (j0 + 1 < DS) ? bias[j0 + 1] : 0.0f;
  sp.z += (j0 + 2 < DS) ? bias[j0 + 2] : 0.0f;
  sp.w  = (j0 + 3 < DS) ? sp.w + bias[j0 + 3] : 0.0f;   // kill pad col 127
  float ss = sp.x * sp.x + sp.y * sp.y + sp.z * sp.z + sp.w * sp.w;
  ss = wave_reduce_sum(ss);
  float un = fmaxf(sqrtf(ss), EPSF);
  float r  = sinhf(un) / un;
  float h0 = sqrtf(1.0f + r * r * ss);
  float* hr = h + (size_t)row * D;
  if (lane == 0)   hr[0]          = h0;
  if (j0 + 0 < DS) hr[1 + j0 + 0] = r * sp.x;
  if (j0 + 1 < DS) hr[1 + j0 + 1] = r * sp.y;
  if (j0 + 2 < DS) hr[1 + j0 + 2] = r * sp.z;
  if (j0 + 3 < DS) hr[1 + j0 + 3] = r * sp.w;
}

// ---- zero d_out (num accumulator)
__global__ void k_zero(float4* __restrict__ p, size_t n4) {
  size_t i = (size_t)blockIdx.x * blockDim.x + threadIdx.x;
  if (i < n4) p[i] = make_float4(0.f, 0.f, 0.f, 0.f);
}

// ---- edge phase: one wave per edge; Minkowski dot, att = 1/(y+sqrt(y^2-1)), scatter-add
__global__ void k_edge(const float* __restrict__ h, const int* __restrict__ ei,
                       float* __restrict__ num, int E) {
  int e = blockIdx.x * 8 + (threadIdx.x >> 5);
  if (e >= E) return;
  int lane = threadIdx.x & 31;
  int r = ei[e];
  int c = ei[(size_t)E + e];
  const float4 xi = *(const float4*)(h + (size_t)r * D + lane * 4);
  const float4 xj = *(const float4*)(h + (size_t)c * D + lane * 4);
  float p = xi.x * xj.x;
  if (lane == 0) p = -p;                       // Minkowski sign on element 0
  p += xi.y * xj.y + xi.z * xj.z + xi.w * xj.w;
  p = wave_reduce_sum(p);                      // l_inner(xi, xj)
  float y   = fmaxf(-p, 1.0f + EPSF);
  float att = 1.0f / (y + sqrtf(y * y - 1.0f));  // exp(-arccosh(y))
  float* o = num + (size_t)r * D + lane * 4;
  atomicAdd(o + 0, att * xj.x);
  atomicAdd(o + 1, att * xj.y);
  atomicAdd(o + 2, att * xj.z);
  atomicAdd(o + 3, att * xj.w);
}

// ---- normalize: out = num / sqrt(max(-l_inner(num,num), EPS)), in place
__global__ void k_norm(float* __restrict__ num, int N) {
  int row = blockIdx.x * 8 + (threadIdx.x >> 5);
  if (row >= N) return;
  int lane = threadIdx.x & 31;
  float4 n4 = *(const float4*)(num + (size_t)row * D + lane * 4);
  float p = n4.x * n4.x;
  if (lane == 0) p = -p;
  p += n4.y * n4.y + n4.z * n4.z + n4.w * n4.w;
  p = wave_reduce_sum(p);                      // l_inner(num, num)
  float inv = 1.0f / sqrtf(fmaxf(-p, EPSF));
  n4.x *= inv; n4.y *= inv; n4.z *= inv; n4.w *= inv;
  *(float4*)(num + (size_t)row * D + lane * 4) = n4;
}

extern "C" void kernel_launch(void* const* d_in, const int* in_sizes, int n_in,
                              void* d_out, int out_size, void* d_ws, size_t ws_size,
                              hipStream_t stream) {
  const float* x    = (const float*)d_in[0];   // (N,128) f32
  const float* w    = (const float*)d_in[1];   // (127,127) f32
  const float* bias = (const float*)d_in[2];   // (127,) f32
  const int*   ei   = (const int*)d_in[3];     // (2,E) int32
  float* out = (float*)d_out;                  // (N,128) f32 — used as num accumulator

  const int N = in_sizes[0] / D;
  const int E = in_sizes[3] / 2;
  const int ntiles = (N + 15) / 16;

  // ws layout: [WhT 32KB][R1: v_h then h][R2: out_space s]
  char* ws = (char*)d_ws;
  _Float16* WhT = (_Float16*)ws;
  const size_t R1  = 32768;
  const size_t RSZ = (size_t)ntiles * 16 * D * sizeof(float);
  _Float16* vh = (_Float16*)(ws + R1);         // N x 128 f16 (dead after k_gemm)
  float*    h  = (float*)(ws + R1);            // N x 128 f32 (overwrites vh region)
  float*    s  = (float*)(ws + R1 + RSZ);      // N x 128 f32

  const int rowBlocks = (N + 7) / 8;           // one wave per row, 8 waves/block
  const size_t n4 = (size_t)out_size / 4;

  k_zero       <<<(int)((n4 + 255) / 256), 256, 0, stream>>>((float4*)out, n4);
  k_prep_weight<<<64, 256, 0, stream>>>(w, WhT);
  k_logmap     <<<rowBlocks, 256, 0, stream>>>(x, vh, N);
  k_gemm       <<<(ntiles + 7) / 8, 256, 0, stream>>>(vh, WhT, s, ntiles);
  k_expmap     <<<rowBlocks, 256, 0, stream>>>(s, bias, h, N);
  k_edge       <<<(E + 7) / 8, 256, 0, stream>>>(h, ei, out, E);
  k_norm       <<<rowBlocks, 256, 0, stream>>>(out, N);
}
